// TFN_11922829214292
// MI455X (gfx1250) — matrix operand order repaired and checked by reading, not compiled, not run on hardware
//
#include <hip/hip_runtime.h>
#include <hip/hip_bf16.h>

// ============================================================================
// SE(3)-equivariant TFN forward for MI455X (gfx1250, wave32, WMMA).
//
// All GEMM-shaped work (radial MLP l1/l2/l3) runs on v_wmma_f32_16x16x32_f16
// with M=16 edge tiles; the big l3 activation tensor R (~245MB/layer if it
// went to HBM) never leaves LDS. Per-edge CG contraction is VALU with
// collision-free (edge,channel) ownership. Segment-sum via
// global_atomic_add_f32. 64-thread blocks (2 waves), 64000B LDS/block.
// Weights are pre-converted to f16 in a TRANSPOSED [col][k] layout so that
// B-fragment loads are contiguous f16x2 (ds_load_b32) and per-chunk LDS
// staging is one contiguous uint4 block copy.
//
// Input layout assumption (jax tree-flatten, dict keys sorted):
//   d_in[0]=t0 (2048x32x1)  d_in[1]=t1 (2048x3x3)  d_in[2]=pos  d_in[3]=w
//   d_in[4..65]   = conv0: 6 rad pairs x {be1,be2,g1,g2,l1.b,l1.w,l2.b,l2.w,
//                   l3.b,l3.w} (keys "0,0","0,1","0,2","1,0","1,1","1,2"),
//                   then self "0", self "1"
//   d_in[66..127] = conv1 (pairs "0,0","0,1","1,0","1,1","2,0","2,1")
//   d_in[128..139]= norm0: d=0,1,2 x {be,g,lin.b,lin.w}
//   d_in[n_in-2]=src  d_in[n_in-1]=dst
// ============================================================================

typedef _Float16 f16;
typedef __attribute__((ext_vector_type(2)))  _Float16 h2v;
typedef __attribute__((ext_vector_type(16))) _Float16 v16h;
typedef __attribute__((ext_vector_type(8)))  float    v8f;

#define TILE 16
#define BLK  64

// ---------------------------------------------------------------------------
// WMMA fragment helpers (wave32; A/B use contiguous k order k = 16h + s, C/D per CDNA5 ISA 7.12.2)
// ---------------------------------------------------------------------------
__device__ inline v8f wmma32(v16h a, v16h b, v8f c) {
  return __builtin_amdgcn_wmma_f32_16x16x32_f16(false, a, false, b,
                                                (short)0, c, false, false);
}

// A: 16x32 f16 tile, row-major in LDS, row stride `ld` (elements, even).
// lanes 0-15: M=lane, K=0..15; lanes 16-31: M=lane-16, K=16..31
// (k,k+1) pairs are contiguous -> 32-bit LDS loads.
__device__ inline v16h frag_load_A(const f16* __restrict__ sA, int ld) {
  const int lane = threadIdx.x & 31;
  const int row  = lane & 15;
  const int kb   = (lane & 16) ? 16 : 0;
  const h2v* pr  = (const h2v*)(sA + row * ld);
  v16h a;
#pragma unroll
  for (int v = 0; v < 8; ++v) {
    const int k = kb + 2 * v;
    h2v t = pr[k >> 1];
    a[2 * v]     = t[0];
    a[2 * v + 1] = t[1];
  }
  return a;
}

// B from TRANSPOSED tile: sBt[col][k], k=0..31 contiguous per column.
// Fragment covers columns [col0, col0+16); lane n: K=0..15, lane 16+n: K=16..31.
__device__ inline v16h frag_load_Bt(const f16* __restrict__ sBt, int col0) {
  const int lane = threadIdx.x & 31;
  const int n    = lane & 15;
  const int kb   = (lane & 16) ? 16 : 0;
  const h2v* pc  = (const h2v*)(sBt + (col0 + n) * 32 + kb);
  v16h b;
#pragma unroll
  for (int v = 0; v < 8; ++v) {
    h2v t = pc[v];
    b[2 * v]     = t[0];
    b[2 * v + 1] = t[1];
  }
  return b;
}

// C/D 16x16 f32: VGPR p: lanes 0-15 -> (M=p, N=lane); lanes 16-31 -> (M=8+p).
__device__ inline void frag_store_C(float* __restrict__ sC, int ld, int colOff,
                                    v8f c) {
  const int lane = threadIdx.x & 31;
  const int n    = lane & 15;
  const int mb   = (lane & 16) ? 8 : 0;
#pragma unroll
  for (int p = 0; p < 8; ++p) sC[(mb + p) * ld + colOff + n] = c[p];
}

__device__ inline void frag_store_C_bias(float* __restrict__ sC, int ld,
                                         int colOff, v8f c, float bias) {
  const int lane = threadIdx.x & 31;
  const int n    = lane & 15;
  const int mb   = (lane & 16) ? 8 : 0;
#pragma unroll
  for (int p = 0; p < 8; ++p) sC[(mb + p) * ld + colOff + n] = c[p] + bias;
}

// contiguous 16B-granule block copy (global -> LDS or LDS -> LDS)
__device__ inline void block_copy16(void* __restrict__ dst,
                                    const void* __restrict__ src, int bytes,
                                    int tid) {
  uint4* d = (uint4*)dst;
  const uint4* s = (const uint4*)src;
  const int n = bytes >> 4;
  for (int i = tid; i < n; i += BLK) d[i] = s[i];
}

// ---------------------------------------------------------------------------
// Q-matrix storage offsets: pairs (di,do) laid out in order
// (0,0)(0,1)(0,2)(1,0)(1,1)(1,2)(2,0)(2,1); within a pair, blocks per J.
// ---------------------------------------------------------------------------
__host__ __device__ static inline int qPairOff(int di, int dd) {
  const int offs[9] = {0, 1, 10, 35, 44, 125, 350, 375, 600};
  return offs[di * 3 + dd];
}

// ---------------------------------------------------------------------------
// Clebsch-Gordan / real-basis Q construction (fp64, tiny, once per launch)
// ---------------------------------------------------------------------------
struct Cplx { double x, y; };
__device__ inline Cplx cmul(Cplx a, Cplx b) {
  return {a.x * b.x - a.y * b.y, a.x * b.y + a.y * b.x};
}
__device__ inline Cplx cconj(Cplx a) { return {a.x, -a.y}; }

__device__ inline double dfact(int n) {
  double r = 1.0;
  for (int i = 2; i <= n; ++i) r *= (double)i;
  return r;
}

__device__ double cgc(int j1, int m1, int j2, int m2, int j3, int m3) {
  if (m1 + m2 != m3) return 0.0;
  double pre = sqrt((2.0 * j3 + 1.0) * dfact(j1 + j2 - j3) * dfact(j1 - j2 + j3) *
                    dfact(-j1 + j2 + j3) / dfact(j1 + j2 + j3 + 1));
  pre *= sqrt(dfact(j1 + m1) * dfact(j1 - m1) * dfact(j2 + m2) * dfact(j2 - m2) *
              dfact(j3 + m3) * dfact(j3 - m3));
  double s = 0.0;
  for (int k = 0; k <= j1 + j2 + j3; ++k) {
    int d0 = k, d1 = j1 + j2 - j3 - k, d2 = j1 - m1 - k, d3 = j2 + m2 - k;
    int d4 = j3 - j2 + m1 + k, d5 = j3 - j1 - m2 + k;
    if (d0 < 0 || d1 < 0 || d2 < 0 || d3 < 0 || d4 < 0 || d5 < 0) continue;
    double t = 1.0 / (dfact(d0) * dfact(d1) * dfact(d2) * dfact(d3) * dfact(d4) *
                      dfact(d5));
    s += (k & 1) ? -t : t;
  }
  return pre * s;
}

__device__ void makeU(int l, Cplx U[9][9]) {
  for (int i = 0; i < 9; ++i)
    for (int j = 0; j < 9; ++j) U[i][j] = {0.0, 0.0};
  U[l][l] = {1.0, 0.0};
  const double s2 = 1.0 / sqrt(2.0);
  for (int m = 1; m <= l; ++m) {
    double sgn = (m & 1) ? -1.0 : 1.0;
    U[l + m][l - m] = {s2, 0.0};
    U[l + m][l + m] = {sgn * s2, 0.0};
    U[l - m][l - m] = {0.0, s2};
    U[l - m][l + m] = {0.0, -sgn * s2};
  }
}

__global__ void build_q_kernel(float* __restrict__ Qbuf) {
  const int p = threadIdx.x;
  if (p >= 8) return;
  const int DIs[8] = {0, 0, 0, 1, 1, 1, 2, 2};
  const int DOs[8] = {0, 1, 2, 0, 1, 2, 0, 1};
  const int di = DIs[p], dd = DOs[p];
  int off = qPairOff(di, dd);
  Cplx Ui[9][9], Uo[9][9], Uj[9][9];
  makeU(di, Ui);
  makeU(dd, Uo);
  const int Jmin = (di > dd) ? (di - dd) : (dd - di);
  const int Jmax = di + dd;
  for (int J = Jmin; J <= Jmax; ++J) {
    makeU(J, Uj);
    const int A = 2 * di + 1, B = 2 * J + 1, Co = 2 * dd + 1;
    double C[5][9][5];
    for (int a = 0; a < 5; ++a)
      for (int b = 0; b < 9; ++b)
        for (int c = 0; c < 5; ++c) C[a][b][c] = 0.0;
    for (int mi = -di; mi <= di; ++mi)
      for (int mJ = -J; mJ <= J; ++mJ) {
        int mo = mi + mJ;
        if (mo < -dd || mo > dd) continue;
        C[di + mi][J + mJ][dd + mo] = cgc(di, mi, J, mJ, dd, mo);
      }
    double Qr[5][5][9], Qi[5][5][9];
    double nr = 0.0, ni = 0.0;
    for (int o = 0; o < Co; ++o)
      for (int i = 0; i < A; ++i)
        for (int j = 0; j < B; ++j) {
          Cplx s = {0.0, 0.0};
          for (int a = 0; a < A; ++a)
            for (int b = 0; b < B; ++b)
              for (int c = 0; c < Co; ++c) {
                double cv = C[a][b][c];
                if (cv == 0.0) continue;
                Cplx t = cmul(Uo[o][c], cmul(cconj(Ui[i][a]), cconj(Uj[j][b])));
                s.x += t.x * cv;
                s.y += t.y * cv;
              }
          Qr[o][i][j] = s.x;
          Qi[o][i][j] = s.y;
          nr += s.x * s.x;
          ni += s.y * s.y;
        }
    const bool useR = (nr >= ni);
    for (int j = 0; j < B; ++j) {
      double cn = 0.0;
      for (int o = 0; o < Co; ++o)
        for (int i = 0; i < A; ++i) {
          double v = useR ? Qr[o][i][j] : Qi[o][i][j];
          cn += v * v;
        }
      cn = sqrt(cn);
      double inv = (cn > 0.0) ? 1.0 / cn : 0.0;
      for (int o = 0; o < Co; ++o)
        for (int i = 0; i < A; ++i) {
          double v = useR ? Qr[o][i][j] : Qi[o][i][j];
          Qbuf[off + (o * A + i) * B + j] = (float)(v * inv);
        }
    }
    off += Co * A * B;
  }
}

// ---------------------------------------------------------------------------
// Weight fp32 -> f16 conversion WITH transpose: dst[col*32+k] = src[k*stride+col]
// (dst writes coalesced; one-time cost per launch, deterministic)
// ---------------------------------------------------------------------------
struct CvtEntry { const float* src; unsigned dstOff; unsigned ncols; unsigned srcStride; };
struct CvtArgs  { CvtEntry e[36]; int count; };

__global__ __launch_bounds__(256) void convert_weights_kernel(CvtArgs a,
                                                              f16* __restrict__ d) {
  for (int j = blockIdx.x; j < a.count; j += gridDim.x) {
    const CvtEntry en = a.e[j];
    const unsigned tot = en.ncols * 32u;
    for (unsigned i = threadIdx.x; i < tot; i += 256) {
      unsigned col = i >> 5, k = i & 31u;
      d[en.dstOff + i] = (f16)en.src[(size_t)k * en.srcStride + col];
    }
  }
}

// ---------------------------------------------------------------------------
// Edge geometry: r, unit vec, real spherical harmonics J<=4, degree count
// ---------------------------------------------------------------------------
__global__ __launch_bounds__(256) void geom_kernel(
    const float* __restrict__ pos, const int* __restrict__ src,
    const int* __restrict__ dst, float* __restrict__ Y, float* __restrict__ rlen,
    float* __restrict__ deg, int E) {
  const int e = blockIdx.x * 256 + threadIdx.x;
  if (e >= E) return;
  const int s = src[e], d = dst[e];
  float dx = pos[d * 3 + 0] - pos[s * 3 + 0];
  float dy = pos[d * 3 + 1] - pos[s * 3 + 1];
  float dz = pos[d * 3 + 2] - pos[s * 3 + 2];
  float r = sqrtf(dx * dx + dy * dy + dz * dz);
  rlen[e] = r;
  float inv = 1.0f / fmaxf(r, 1e-8f);
  float x = dx * inv, y = dy * inv, z = dz * inv;
  float phi = atan2f(y, x);
  float ct = fminf(fmaxf(z, -1.f), 1.f);
  float st = sqrtf(fmaxf(1.f - ct * ct, 0.f));
  float P[5][5];
  float pmm = 1.f;
  for (int m = 0; m <= 4; ++m) {
    if (m > 0) pmm *= -(2.f * m - 1.f) * st;
    P[m][m] = pmm;
    if (m + 1 <= 4) P[m + 1][m] = ct * (2.f * m + 1.f) * pmm;
    for (int l = m + 2; l <= 4; ++l)
      P[l][m] = ((2.f * l - 1.f) * ct * P[l - 1][m] - (l + m - 1.f) * P[l - 2][m]) /
                (float)(l - m);
  }
  const float fourpi = 12.566370614359172f;
  const float sq2 = 1.41421356237309515f;
  for (int J = 0; J <= 4; ++J) {
    for (int m = -J; m <= J; ++m) {
      int am = (m < 0) ? -m : m;
      float fr = 1.f;
      for (int i = J - am + 1; i <= J + am; ++i) fr *= (float)i;  // (J+am)!/(J-am)!
      float K = sqrtf((2.f * J + 1.f) / fourpi / fr);
      float sgn = (am & 1) ? -1.f : 1.f;
      float v;
      if (m == 0)      v = K * P[J][0];
      else if (m > 0)  v = sq2 * sgn * K * cosf((float)m * phi) * P[J][am];
      else             v = sq2 * sgn * K * sinf((float)am * phi) * P[J][am];
      Y[(size_t)e * 25 + J * J + (m + J)] = v;
    }
  }
  atomicAdd(&deg[d], 1.0f);
}

__global__ __launch_bounds__(256) void invdeg_kernel(const float* __restrict__ deg,
                                                     float* __restrict__ invd,
                                                     int n) {
  int i = blockIdx.x * 256 + threadIdx.x;
  if (i < n) {
    float d = deg[i];
    invd[i] = (d > 0.f) ? 1.f / fmaxf(d, 1.f) : 0.f;
  }
}

// ---------------------------------------------------------------------------
// Fused equivariant conv layer (WMMA radial MLPs + CG contraction + scatter)
// ---------------------------------------------------------------------------
struct PairA {
  const f16* w1h;      // 32x32 f16, TRANSPOSED [col][k] (rows 0..31 of l1.w)
  const f16* w2h;      // 32x32 f16, TRANSPOSED
  const f16* w3h;      // ncols x 32 f16, TRANSPOSED [col][k]
  const float* w1row32;  // l1.w row 32 (r feature), fp32
  const float* b1; const float* g1; const float* be1;
  const float* b2; const float* g2; const float* be2;
  const float* b3;
  int di, nci, nJ, ncols;
};
struct ConvA {
  int n_do, n_di;
  int dd[3], nco[3];
  const float* selfw[3];
  int snci[3];
  const float* hdst[3];
  float* agg[3];
  const float* hin[3];
  PairA pair[3][3];  // [out-degree idx][in-degree idx]
};

// LDS layout (bytes); total 64000 < 64KB, all offsets 64B aligned
constexpr int SM_RF    = 0;                 // 16x34 f16   = 1088
constexpr int SM_RLEN  = SM_RF + 1088;      // 16 f32      = 64
constexpr int SM_Y     = SM_RLEN + 64;      // 16x25 f32   = 1600
constexpr int SM_SRC   = SM_Y + 1600;       // 16 int      = 64
constexpr int SM_DST   = SM_SRC + 64;       // 16 int      = 64
constexpr int SM_PAR   = SM_DST + 64;       // 7x32 f32    = 896 (b1,g1,be1,b2,g2,be2,w1row32)
constexpr int SM_H1    = SM_PAR + 896;      // 16x32 f16   = 1024
constexpr int SM_H2    = SM_H1 + 1024;      // 16x32 f16   = 1024
constexpr int SM_HS    = SM_H2 + 1024;      // 16x160 f32  = 10240
constexpr int SM_BAS   = SM_HS + 10240;     // 16x45 f32   = 2880
constexpr int SM_T     = SM_BAS + 2880;     // 16x288 f32  = 18432
constexpr int SM_R     = SM_T + 18432;      // 16x128 f32  = 8192 (also selfw stage)
constexpr int SM_MSG   = SM_R + 8192;       // 16x160 f32  = 10240
constexpr int SM_UNION = SM_MSG + 10240;    // 8192 (sC | sW3 | sHD; +sWB at +2048)
constexpr int SM_TOTAL = SM_UNION + 8192;

// branch-free LN+ReLU fixup; params come from LDS-staged block
__device__ inline void ln_relu_to_f16(const float* __restrict__ sC,
                                      const float* __restrict__ b,
                                      const float* __restrict__ g,
                                      const float* __restrict__ be,
                                      const float* __restrict__ ew,
                                      float xr, f16* __restrict__ outH, int tid) {
  if (tid < TILE) {
    float v[32];
    float mu = 0.f;
#pragma unroll 8
    for (int n = 0; n < 32; ++n) {
      float t = sC[tid * 32 + n] + b[n] + xr * ew[n];
      v[n] = t;
      mu += t;
    }
    mu *= (1.f / 32.f);
    float var = 0.f;
#pragma unroll 8
    for (int n = 0; n < 32; ++n) {
      float d = v[n] - mu;
      var += d * d;
    }
    var *= (1.f / 32.f);
    float iv = rsqrtf(var + 1e-5f);
#pragma unroll 8
    for (int n = 0; n < 32; ++n) {
      float t = (v[n] - mu) * iv * g[n] + be[n];
      outH[tid * 32 + n] = (f16)fmaxf(t, 0.f);
    }
  }
}

__global__ __launch_bounds__(BLK) void conv_kernel(
    ConvA A, const int* __restrict__ src, const int* __restrict__ dst,
    const float* __restrict__ wfeat, const float* __restrict__ rlen,
    const float* __restrict__ Yall, const float* __restrict__ Qbuf) {
  __shared__ unsigned char smem[SM_TOTAL] __attribute__((aligned(16)));
  f16*   sRF  = (f16*)(smem + SM_RF);
  float* sRL  = (float*)(smem + SM_RLEN);
  float* sY   = (float*)(smem + SM_Y);
  int*   sSrc = (int*)(smem + SM_SRC);
  int*   sDst = (int*)(smem + SM_DST);
  float* sPar = (float*)(smem + SM_PAR);
  f16*   sH1  = (f16*)(smem + SM_H1);
  f16*   sH2  = (f16*)(smem + SM_H2);
  float* sHS  = (float*)(smem + SM_HS);
  float* sBas = (float*)(smem + SM_BAS);
  float* sT   = (float*)(smem + SM_T);
  float* sR   = (float*)(smem + SM_R);
  float* sMsg = (float*)(smem + SM_MSG);
  float* sC   = (float*)(smem + SM_UNION);            // l1/l2 phase
  f16*   sWB  = (f16*)(smem + SM_UNION + 2048);       // l1/l2 phase
  f16*   sW3  = (f16*)(smem + SM_UNION);              // l3 phase
  float* sHD  = (float*)(smem + SM_UNION);            // self phase
  float* sSW  = sR;                                   // self phase (sR free)

  const int tid  = threadIdx.x;
  const int wave = tid >> 5;
  const int tile = blockIdx.x * TILE;

  if (tid < TILE) {
    sSrc[tid] = src[tile + tid];
    sDst[tid] = dst[tile + tid];
    float r = rlen[tile + tid];
    sRL[tid] = r;
    sRF[tid * 34 + 32] = (f16)r;
    sRF[tid * 34 + 33] = (f16)0.f;
  }
  for (int i = tid; i < TILE * 32; i += BLK) {
    int e = i >> 5, n = i & 31;
    sRF[e * 34 + n] = (f16)wfeat[(size_t)(tile + e) * 32 + n];
  }
  for (int i = tid; i < TILE * 25; i += BLK) {
    int e = i / 25, j = i - e * 25;
    sY[e * 25 + j] = Yall[(size_t)(tile + e) * 25 + j];
  }
  __syncthreads();

  for (int oi = 0; oi < A.n_do; ++oi) {
    const int dd = A.dd[oi], nco = A.nco[oi], W = 2 * dd + 1;
    for (int i = tid; i < TILE * nco * W; i += BLK) sMsg[i] = 0.f;
    __syncthreads();

    for (int ii = 0; ii < A.n_di; ++ii) {
      const PairA P = A.pair[oi][ii];
      const int di = P.di, nci = P.nci, V = 2 * di + 1, nJ = P.nJ, ncols = P.ncols;
      const int Jmin = (di > dd) ? (di - dd) : (dd - di);

      // gather h_src[e][c][iv]
      for (int i = tid; i < TILE * nci * V; i += BLK) {
        int e = i / (nci * V), q = i - e * nci * V;
        sHS[i] = A.hin[ii][(size_t)sSrc[e] * nci * V + q];
      }
      // basis[e][o][iv][f] = sum_j Y[e, J=Jmin+f, j] * Q[o,iv,j]
      int qoff[3];
      {
        int a = 0;
        for (int f = 0; f < nJ; ++f) { qoff[f] = a; a += W * V * (2 * (Jmin + f) + 1); }
      }
      const float* Qp = Qbuf + qPairOff(di, dd);
      for (int i = tid; i < TILE * W * V * nJ; i += BLK) {
        int f = i % nJ, t = i / nJ;
        int iv = t % V; t /= V;
        int o = t % W, e = t / W;
        int J = Jmin + f, Jd = 2 * J + 1;
        float a = 0.f;
        for (int j = 0; j < Jd; ++j)
          a += sY[e * 25 + J * J + j] * Qp[qoff[f] + (o * V + iv) * Jd + j];
        sBas[((e * W + o) * V + iv) * nJ + f] = a;
      }
      // stage LN params (7x32) + W1 (transposed 32x32 f16)
      {
        const float* ps[7] = {P.b1, P.g1, P.be1, P.b2, P.g2, P.be2, P.w1row32};
        for (int i = tid; i < 224; i += BLK) sPar[i] = ps[i >> 5][i & 31];
      }
      block_copy16(sWB, P.w1h, 2048, tid);
      __syncthreads();

      // ---- radial l1: WMMA(rfeat 16x32, W1 32x32) + r-row + bias + LN + ReLU
      {
        v16h a = frag_load_A(sRF, 34);
        v16h b = frag_load_Bt(sWB, wave * 16);
        v8f  c = {0.f, 0.f, 0.f, 0.f, 0.f, 0.f, 0.f, 0.f};
        c = wmma32(a, b, c);
        frag_store_C(sC, 32, wave * 16, c);
      }
      __syncthreads();
      ln_relu_to_f16(sC, sPar + 0, sPar + 32, sPar + 64, sPar + 192,
                     (tid < TILE) ? sRL[tid] : 0.f, sH1, tid);
      __syncthreads();

      // ---- radial l2
      block_copy16(sWB, P.w2h, 2048, tid);
      __syncthreads();
      {
        v16h a = frag_load_A(sH1, 32);
        v16h b = frag_load_Bt(sWB, wave * 16);
        v8f  c = {0.f, 0.f, 0.f, 0.f, 0.f, 0.f, 0.f, 0.f};
        c = wmma32(a, b, c);
        frag_store_C(sC, 32, wave * 16, c);
      }
      __syncthreads();
      ln_relu_to_f16(sC, sPar + 96, sPar + 128, sPar + 160, sPar + 192, 0.f,
                     sH2, tid);
      __syncthreads();

      // ---- T[e][c][o][f] = sum_iv basis[e,o,iv,f] * hs[e,c,iv]
      for (int i = tid; i < TILE * nci * W * nJ; i += BLK) {
        int f = i % nJ, t = i / nJ;
        int o = t % W; t /= W;
        int c = t % nci, e = t / nci;
        float acc = 0.f;
        for (int iv = 0; iv < V; ++iv)
          acc += sBas[((e * W + o) * V + iv) * nJ + f] * sHS[(e * nci + c) * V + iv];
        sT[((e * nci + c) * W + o) * nJ + f] = acc;
      }
      __syncthreads();

      // ---- l3 in 128-col chunks (WMMA, bias folded) + collision-free contraction
      const int g = nci * nJ;
      v16h aH2 = frag_load_A(sH2, 32);
      for (int c0 = 0; c0 < ncols; c0 += 128) {
        const int cc = (ncols - c0 < 128) ? (ncols - c0) : 128;
        // chunk of transposed W3 is one contiguous block: cols [c0,c0+cc) x 32
        block_copy16(sW3, P.w3h + (size_t)c0 * 32, cc * 64, tid);
        __syncthreads();
        for (int n0 = wave * 16; n0 < cc; n0 += 32) {
          v16h b = frag_load_Bt(sW3, n0);
          v8f  c = {0.f, 0.f, 0.f, 0.f, 0.f, 0.f, 0.f, 0.f};
          c = wmma32(aH2, b, c);
          float bias = P.b3[c0 + n0 + (tid & 15)];
          frag_store_C_bias(sR, 128, n0, c, bias);
        }
        __syncthreads();
        // each thread owns (edge e, out-channel k): no write collisions
        for (int slot = tid; slot < TILE * nco; slot += BLK) {
          int e = slot / nco, k = slot - e * nco;
          int lo = k * g;       if (c0 > lo) lo = c0;
          int hi = (k + 1) * g; int ce = c0 + cc; if (ce < hi) hi = ce;
          for (int col = lo; col < hi; ++col) {
            int q = col - k * g;
            int cI = q / nJ, f = q - cI * nJ;
            float rv = sR[e * 128 + (col - c0)];
            float* mrow = &sMsg[(e * nco + k) * W];
            const float* trow = &sT[((e * nci + cI) * W) * nJ + f];
            for (int o = 0; o < W; ++o) mrow[o] += rv * trow[o * nJ];
          }
        }
        __syncthreads();
      }
    }

    // ---- self interaction: msg[e,k,o] += sum_c selfw[k,c]*h[do][dst[e],c,o]
    if (A.selfw[oi]) {
      const int snci = A.snci[oi];
      for (int i = tid; i < TILE * snci * W; i += BLK) {
        int e = i / (snci * W), q = i - e * snci * W;
        sHD[i] = A.hdst[oi][(size_t)sDst[e] * snci * W + q];
      }
      for (int i = tid; i < nco * snci; i += BLK) sSW[i] = A.selfw[oi][i];
      __syncthreads();
      for (int slot = tid; slot < TILE * nco; slot += BLK) {
        int e = slot / nco, k = slot - e * nco;
        for (int o = 0; o < W; ++o) {
          float acc = 0.f;
          for (int c = 0; c < snci; ++c)
            acc += sSW[k * snci + c] * sHD[(e * snci + c) * W + o];
          sMsg[(e * nco + k) * W + o] += acc;
        }
      }
      __syncthreads();
    }

    // ---- scatter (segment sum)
    float* aggp = A.agg[oi];
    for (int i = tid; i < TILE * nco * W; i += BLK) {
      int e = i / (nco * W), q = i - e * (nco * W);
      atomicAdd(&aggp[(size_t)sDst[e] * nco * W + q], sMsg[i]);
    }
    __syncthreads();
  }
}

// ---------------------------------------------------------------------------
// Equivariant norm (one block of 32 threads per node)
// ---------------------------------------------------------------------------
__device__ inline float wave_sum32(float v) {
  for (int m = 16; m; m >>= 1) v += __shfl_xor(v, m, 32);
  return v;
}

__global__ __launch_bounds__(32) void gnorm_kernel(
    const float* __restrict__ agg, const float* __restrict__ invd,
    float* __restrict__ out, const float* __restrict__ g,
    const float* __restrict__ be, const float* __restrict__ lw,
    const float* __restrict__ lb, int W) {
  const int n = blockIdx.x, c = threadIdx.x;
  float v[5];
  float id = invd[n];
  float ss = 0.f;
  for (int o = 0; o < W; ++o) {
    float t = agg[((size_t)n * 32 + c) * W + o] * id;
    v[o] = t;
    ss += t * t;
  }
  float nrm = sqrtf(ss);
  float mu = wave_sum32(nrm) * (1.f / 32.f);
  float d0 = nrm - mu;
  float var = wave_sum32(d0 * d0) * (1.f / 32.f);
  float t = d0 * rsqrtf(var + 1e-5f) * g[c] + be[c];
  t = fmaxf(t, 0.f);
  __shared__ float sT[32];
  sT[c] = t;
  __syncthreads();
  float acc = lb[c];
  for (int cc = 0; cc < 32; ++cc) acc += sT[cc] * lw[cc * 32 + c];
  float den = fmaxf(nrm, 1e-12f);
  for (int o = 0; o < W; ++o)
    out[((size_t)n * 32 + c) * W + o] = acc * (v[o] / den);
}

__global__ __launch_bounds__(256) void finalize_kernel(
    const float* __restrict__ a0, const float* __restrict__ a1,
    const float* __restrict__ invd, float* __restrict__ out, int N) {
  int i = blockIdx.x * 256 + threadIdx.x;
  int n0 = N * 32;
  if (i < n0) {
    out[i] = a0[i] * invd[i >> 5];
  } else if (i < n0 + N * 9) {
    int j = i - n0;
    out[i] = a1[j] * invd[j / 9];
  }
}

// ---------------------------------------------------------------------------
// Host launch
// ---------------------------------------------------------------------------
struct PD { int di, dd, nci, nco, nJ; };

extern "C" void kernel_launch(void* const* d_in, const int* in_sizes, int n_in,
                              void* d_out, int out_size, void* d_ws, size_t ws_size,
                              hipStream_t stream) {
  (void)out_size; (void)ws_size;
  const int E = in_sizes[n_in - 1];   // 16384 edges
  const int N = in_sizes[2] / 3;      // 2048 nodes

  const float* t0  = (const float*)d_in[0];
  const float* t1  = (const float*)d_in[1];
  const float* pos = (const float*)d_in[2];
  const float* wf  = (const float*)d_in[3];
  const int* src = (const int*)d_in[n_in - 2];
  const int* dst = (const int*)d_in[n_in - 1];

  // workspace carve
  unsigned char* wsb = (unsigned char*)d_ws;
  size_t off = 0;
  auto carve = [&](size_t bytes) -> void* {
    void* p = wsb + off;
    off = (off + bytes + 255) & ~((size_t)255);
    return p;
  };
  float* Qbuf = (float*)carve(600 * 4);
  float* Yb   = (float*)carve((size_t)E * 25 * 4);
  float* rl   = (float*)carve((size_t)E * 4);
  float* deg  = (float*)carve((size_t)N * 4);
  float* invd = (float*)carve((size_t)N * 4);
  float* agg0[3] = {(float*)carve((size_t)N * 32 * 4),
                    (float*)carve((size_t)N * 96 * 4),
                    (float*)carve((size_t)N * 160 * 4)};
  float* hmid[3] = {(float*)carve((size_t)N * 32 * 4),
                    (float*)carve((size_t)N * 96 * 4),
                    (float*)carve((size_t)N * 160 * 4)};
  float* agg1[2] = {(float*)carve((size_t)N * 32 * 4),
                    (float*)carve((size_t)N * 9 * 4)};
  f16* hb = (f16*)carve((size_t)270000 * 2);

  static const PD L0P[6] = {{0, 0, 32, 32, 1}, {0, 1, 32, 32, 1}, {0, 2, 32, 32, 1},
                            {1, 0, 3, 32, 1},  {1, 1, 3, 32, 3},  {1, 2, 3, 32, 3}};
  static const PD L1P[6] = {{0, 0, 32, 32, 1}, {0, 1, 32, 3, 1},  {1, 0, 32, 32, 1},
                            {1, 1, 32, 3, 3},  {2, 0, 32, 32, 1}, {2, 1, 32, 3, 3}};

  CvtArgs cv{};
  cv.count = 36;
  const f16* w1h[2][6];
  const f16* w2h[2][6];
  const f16* w3h[2][6];
  size_t hoff = 0;
  int ce = 0;
  for (int L = 0; L < 2; ++L) {
    const PD* PDs = L ? L1P : L0P;
    const int base0 = L ? 66 : 4;
    for (int pI = 0; pI < 6; ++pI) {
      const int b = base0 + pI * 10;
      const unsigned ncols = (unsigned)(PDs[pI].nco * PDs[pI].nci * PDs[pI].nJ);
      // w1: take rows 0..31 of (33x32), transpose -> [32 cols][32 k]
      cv.e[ce++] = {(const float*)d_in[b + 5], (unsigned)hoff, 32u, 32u};
      w1h[L][pI] = hb + hoff; hoff += 1024;
      cv.e[ce++] = {(const float*)d_in[b + 7], (unsigned)hoff, 32u, 32u};
      w2h[L][pI] = hb + hoff; hoff += 1024;
      cv.e[ce++] = {(const float*)d_in[b + 9], (unsigned)hoff, ncols, ncols};
      w3h[L][pI] = hb + hoff; hoff += (size_t)32 * ncols;
    }
  }

  auto fillPair = [&](PairA& P, int L, int pI, int base0) {
    const PD* PDs = L ? L1P : L0P;
    const int b = base0 + pI * 10;
    const PD& q = PDs[pI];
    P.w1h = w1h[L][pI]; P.w2h = w2h[L][pI]; P.w3h = w3h[L][pI];
    P.w1row32 = ((const float*)d_in[b + 5]) + 32 * 32;
    P.b1 = (const float*)d_in[b + 4]; P.g1 = (const float*)d_in[b + 2];
    P.be1 = (const float*)d_in[b + 0];
    P.b2 = (const float*)d_in[b + 6]; P.g2 = (const float*)d_in[b + 3];
    P.be2 = (const float*)d_in[b + 1];
    P.b3 = (const float*)d_in[b + 8];
    P.di = q.di; P.nci = q.nci; P.nJ = q.nJ; P.ncols = q.nco * q.nci * q.nJ;
  };

  ConvA A0{};
  A0.n_do = 3; A0.n_di = 2;
  for (int oi = 0; oi < 3; ++oi) { A0.dd[oi] = oi; A0.nco[oi] = 32; A0.agg[oi] = agg0[oi]; }
  A0.selfw[0] = (const float*)d_in[64]; A0.snci[0] = 32; A0.hdst[0] = t0;
  A0.selfw[1] = (const float*)d_in[65]; A0.snci[1] = 3;  A0.hdst[1] = t1;
  A0.selfw[2] = nullptr; A0.snci[2] = 0; A0.hdst[2] = nullptr;
  A0.hin[0] = t0; A0.hin[1] = t1; A0.hin[2] = nullptr;
  for (int oi = 0; oi < 3; ++oi)
    for (int ii = 0; ii < 2; ++ii)
      fillPair(A0.pair[oi][ii], 0, ii * 3 + oi, 4);

  ConvA A1{};
  A1.n_do = 2; A1.n_di = 3;
  A1.dd[0] = 0; A1.nco[0] = 32; A1.agg[0] = agg1[0];
  A1.dd[1] = 1; A1.nco[1] = 3;  A1.agg[1] = agg1[1];
  A1.selfw[0] = (const float*)d_in[126]; A1.snci[0] = 32; A1.hdst[0] = hmid[0];
  A1.selfw[1] = (const float*)d_in[127]; A1.snci[1] = 32; A1.hdst[1] = hmid[1];
  A1.selfw[2] = nullptr; A1.snci[2] = 0; A1.hdst[2] = nullptr;
  A1.hin[0] = hmid[0]; A1.hin[1] = hmid[1]; A1.hin[2] = hmid[2];
  for (int oi = 0; oi < 2; ++oi)
    for (int ii = 0; ii < 3; ++ii)
      fillPair(A1.pair[oi][ii], 1, ii * 2 + oi, 66);

  (void)hipMemsetAsync(deg, 0, (size_t)N * 4, stream);
  (void)hipMemsetAsync(agg0[0], 0, (size_t)N * 32 * 4, stream);
  (void)hipMemsetAsync(agg0[1], 0, (size_t)N * 96 * 4, stream);
  (void)hipMemsetAsync(agg0[2], 0, (size_t)N * 160 * 4, stream);
  (void)hipMemsetAsync(agg1[0], 0, (size_t)N * 32 * 4, stream);
  (void)hipMemsetAsync(agg1[1], 0, (size_t)N * 9 * 4, stream);

  build_q_kernel<<<1, 8, 0, stream>>>(Qbuf);
  convert_weights_kernel<<<36, 256, 0, stream>>>(cv, hb);
  geom_kernel<<<(E + 255) / 256, 256, 0, stream>>>(pos, src, dst, Yb, rl, deg, E);
  invdeg_kernel<<<(N + 255) / 256, 256, 0, stream>>>(deg, invd, N);

  conv_kernel<<<E / TILE, BLK, 0, stream>>>(A0, src, dst, wf, rl, Yb, Qbuf);

  for (int d = 0; d < 3; ++d) {
    const int b = 128 + d * 4;  // be, g, lin.b, lin.w
    gnorm_kernel<<<N, 32, 0, stream>>>(agg0[d], invd, hmid[d],
                                       (const float*)d_in[b + 1],
                                       (const float*)d_in[b + 0],
                                       (const float*)d_in[b + 3],
                                       (const float*)d_in[b + 2], 2 * d + 1);
  }

  conv_kernel<<<E / TILE, BLK, 0, stream>>>(A1, src, dst, wf, rl, Yb, Qbuf);

  finalize_kernel<<<(N * 41 + 255) / 256, 256, 0, stream>>>(agg1[0], agg1[1], invd,
                                                            (float*)d_out, N);
}
